// Attention_3985729650994
// MI455X (gfx1250) — matrix operand order repaired and checked by reading, not compiled, not run on hardware
//
#include <hip/hip_runtime.h>

// ---------------------------------------------------------------------------
// MHA forward for B=2, S=2048, D=1024, H=16, hd=64 on gfx1250 (MI455X).
// bf16 WMMA (v_wmma_f32_16x16x32_bf16), fp32 accumulate/softmax.
// Tiles staged with GLOBAL_LOAD_ASYNC_TO_LDS_B128 + double-buffered LDS so
// the async copy of tile i+1 overlaps the WMMAs of tile i (ASYNCcnt path).
// ---------------------------------------------------------------------------

typedef __bf16  bf16_t;
typedef bf16_t  v16bf __attribute__((ext_vector_type(16)));
typedef bf16_t  v8bf  __attribute__((ext_vector_type(8)));
typedef float   v8f   __attribute__((ext_vector_type(8)));
typedef int     v4i   __attribute__((ext_vector_type(4)));

#define BATCH   2
#define SEQ     2048
#define DIM     1024
#define NHEAD   16
#define HD      64
#define MROWS   (BATCH * SEQ)      // 4096

union FragBF {
    v16bf v;
    v8bf  h[2];
};

// --------------------------- async LDS staging -----------------------------
#if defined(__has_builtin)
#if __has_builtin(__builtin_amdgcn_global_load_async_to_lds_b128)
#define HAVE_ASYNC_LDS 1
#endif
#endif

__device__ __forceinline__ void copy16_async(const bf16_t* g, bf16_t* l) {
#ifdef HAVE_ASYNC_LDS
    // Signature (from hipcc diagnostic): (v4i AS1*, v4i AS3*, imm off, imm cpol)
    __builtin_amdgcn_global_load_async_to_lds_b128(
        (__attribute__((address_space(1))) v4i*)g,
        (__attribute__((address_space(3))) v4i*)l, 0, 0);
#else
    *(uint4*)l = *(const uint4*)g;
#endif
}

__device__ __forceinline__ void wait_async() {
#ifdef HAVE_ASYNC_LDS
    asm volatile("s_wait_asynccnt 0" ::: "memory");
#endif
}

// ---------------------------------------------------------------------------
// fp32 -> bf16 conversion (vectorized by 4)
// ---------------------------------------------------------------------------
__global__ void cvt_f32_to_bf16(const float* __restrict__ src,
                                bf16_t* __restrict__ dst, int n4) {
    int i = blockIdx.x * blockDim.x + threadIdx.x;
    if (i < n4) {
        float4 v = ((const float4*)src)[i];
        dst[i * 4 + 0] = (bf16_t)v.x;
        dst[i * 4 + 1] = (bf16_t)v.y;
        dst[i * 4 + 2] = (bf16_t)v.z;
        dst[i * 4 + 3] = (bf16_t)v.w;
    }
}

// ---------------------------------------------------------------------------
// Tiled GEMM: C[M,N] = A[M,K] * W[N,K]^T   (torch Linear convention)
// block = 128 threads (4 wave32), C tile = 64x64, K-chunk = 64,
// double-buffered LDS, async global->LDS staging.
// Each wave computes a 16x64 strip: 4 accumulators, 8 WMMAs per K-chunk.
// mode 0/1: bf16 out scattered to per-head [b,h,s,hd] (mode 0 applies scale)
// mode 2  : bf16 out scattered to transposed per-head [b,h,hd,s]
// mode 3  : fp32 out = acc + bias[n] at [m,n] (flat)
// ---------------------------------------------------------------------------
__global__ __launch_bounds__(128)
void gemm_bf16(const bf16_t* __restrict__ A, const bf16_t* __restrict__ W,
               void* __restrict__ outp, const float* __restrict__ bias,
               int Kdim, int Ndim, int mode, float scale) {
    __shared__ alignas(16) bf16_t ldsA[2][64][64];
    __shared__ alignas(16) bf16_t ldsB[2][64][64];

    const int m0   = blockIdx.x * 64;
    const int n0   = blockIdx.y * 64;
    const int t    = threadIdx.x;
    const int wave = t >> 5;
    const int lane = t & 31;
    const int lr   = lane & 15;   // row/col within 16-strip
    const int hi   = lane >> 4;   // half-wave select

    v8f acc[4] = {};

    // cooperative tile copy: 2 threads per 64-elem row, 32 elems each
    const int cr = t >> 1;
    const int cc = (t & 1) * 32;

    auto issue = [&](int k0, int bufw) {
        const bf16_t* ga = A + (size_t)(m0 + cr) * Kdim + k0 + cc;
        bf16_t* la = &ldsA[bufw][cr][cc];
        const bf16_t* gw = W + (size_t)(n0 + cr) * Kdim + k0 + cc;
        bf16_t* lb = &ldsB[bufw][cr][cc];
#pragma unroll
        for (int q = 0; q < 4; ++q) {
            copy16_async(ga + q * 8, la + q * 8);
            copy16_async(gw + q * 8, lb + q * 8);
        }
    };

    const int nIter = Kdim >> 6;
    issue(0, 0);
    for (int it = 0; it < nIter; ++it) {
        wait_async();
        __syncthreads();                       // tile `it` visible to all waves
        if (it + 1 < nIter) issue((it + 1) << 6, (it + 1) & 1);
        const int buf = it & 1;

        // A fragments (16x32 each) for this wave's 16-row strip
        FragBF af[2];
#pragma unroll
        for (int kk = 0; kk < 2; ++kk) {
            const bf16_t* ar = &ldsA[buf][wave * 16 + lr][kk * 32];
            af[kk].h[0] = *(const v8bf*)(ar + hi * 8);
            af[kk].h[1] = *(const v8bf*)(ar + 16 + hi * 8);
        }
#pragma unroll
        for (int kk = 0; kk < 2; ++kk) {
            FragBF bfr[4];                     // load all B frags, then 4 WMMAs
#pragma unroll
            for (int g = 0; g < 4; ++g) {
                const bf16_t* br = &ldsB[buf][g * 16 + lr][kk * 32 + hi * 8];
                bfr[g].h[0] = *(const v8bf*)(br);
                bfr[g].h[1] = *(const v8bf*)(br + 16);
            }
#pragma unroll
            for (int g = 0; g < 4; ++g)
                acc[g] = __builtin_amdgcn_wmma_f32_16x16x32_bf16(
                    false, af[kk].v, false, bfr[g].v, (short)0, acc[g], false, false);
        }
        __syncthreads();                       // reads done before buf reuse
    }

    // Epilogue. C layout: element j of a lane = row (hi*8+j), col (lane&15).
#pragma unroll
    for (int g = 0; g < 4; ++g) {
#pragma unroll
        for (int j = 0; j < 8; ++j) {
            const int m = m0 + wave * 16 + hi * 8 + j;
            const int n = n0 + g * 16 + lr;
            float v = acc[g][j] * scale;
            if (mode == 3) {
                ((float*)outp)[(size_t)m * Ndim + n] = v + bias[n];
            } else {
                const int b = m >> 11, s = m & (SEQ - 1);
                const int h = n >> 6, d = n & (HD - 1);
                bf16_t* o = (bf16_t*)outp;
                if (mode == 2)
                    o[(((size_t)(b * NHEAD + h)) * HD + d) * SEQ + s] = (bf16_t)v;
                else
                    o[(((size_t)(b * NHEAD + h)) * SEQ + s) * HD + d] = (bf16_t)v;
            }
        }
    }
}

// ---------------------------------------------------------------------------
// Flash-style attention. grid = (B*H, S/64), block = 128 (4 waves).
// Each block: 64 query rows of one (b,h). Streams 64-key tiles of K and V^T
// through double-buffered LDS (async staging); online softmax in fp32;
// P -> A-layout via per-wave LDS buffer. Q is pre-scaled by 1/sqrt(hd).
// ---------------------------------------------------------------------------
__global__ __launch_bounds__(128)
void attn_kernel(const bf16_t* __restrict__ Q, const bf16_t* __restrict__ Kb,
                 const bf16_t* __restrict__ Vt, bf16_t* __restrict__ O) {
    __shared__ alignas(16) bf16_t ldsK[2][64][64];     // [key][hd]
    __shared__ alignas(16) bf16_t ldsV[2][64][64];     // [hd][key]
    __shared__ alignas(16) bf16_t ldsP[4][16][64];     // per-wave P strip

    const int bh   = blockIdx.x;               // 0..31
    const int q0   = blockIdx.y * 64;
    const int bidx = bh >> 4;
    const int head = bh & (NHEAD - 1);
    const int t    = threadIdx.x;
    const int wave = t >> 5;
    const int lane = t & 31;
    const int lr   = lane & 15;
    const int hi   = lane >> 4;

    const size_t headOff = (size_t)bh * SEQ * HD;  // same for Q, K, Vt

    // Load Q strip (16 rows x 64) as two A fragments, kept in registers.
    FragBF qf[2];
    const bf16_t* qrow = Q + headOff + (size_t)(q0 + wave * 16 + lr) * HD;
#pragma unroll
    for (int kk = 0; kk < 2; ++kk) {
        qf[kk].h[0] = *(const v8bf*)(qrow + kk * 32 + hi * 8);
        qf[kk].h[1] = *(const v8bf*)(qrow + kk * 32 + 16 + hi * 8);
    }

    float mrow[8], lsum[8];
    v8f o[4] = {};
#pragma unroll
    for (int j = 0; j < 8; ++j) { mrow[j] = -1e30f; lsum[j] = 0.f; }

    const int cr = t >> 1;
    const int cc = (t & 1) * 32;
    const bf16_t* Kh = Kb + headOff;
    const bf16_t* Vh = Vt + headOff;

    auto issueTiles = [&](int kb, int bufw) {
        const bf16_t* gk = Kh + (size_t)(kb + cr) * HD + cc;
        bf16_t* lk = &ldsK[bufw][cr][cc];
        const bf16_t* gv = Vh + (size_t)cr * SEQ + kb + cc;
        bf16_t* lv = &ldsV[bufw][cr][cc];
#pragma unroll
        for (int q = 0; q < 4; ++q) {
            copy16_async(gk + q * 8, lk + q * 8);
            copy16_async(gv + q * 8, lv + q * 8);
        }
    };

    const int nIter = SEQ / 64;                // 32
    issueTiles(0, 0);
    for (int it = 0; it < nIter; ++it) {
        wait_async();
        __syncthreads();
        if (it + 1 < nIter) issueTiles((it + 1) * 64, (it + 1) & 1);
        const int buf = it & 1;

        // S strip = Q(16x64) * K_tile^T -> 4 tiles of 16x16
        v8f s[4] = {};
#pragma unroll
        for (int kk = 0; kk < 2; ++kk) {
            FragBF kf[4];
#pragma unroll
            for (int g = 0; g < 4; ++g) {
                const bf16_t* kr = &ldsK[buf][g * 16 + lr][kk * 32 + hi * 8];
                kf[g].h[0] = *(const v8bf*)(kr);
                kf[g].h[1] = *(const v8bf*)(kr + 16);
            }
#pragma unroll
            for (int g = 0; g < 4; ++g)
                s[g] = __builtin_amdgcn_wmma_f32_16x16x32_bf16(
                    false, qf[kk].v, false, kf[g].v, (short)0, s[g], false, false);
        }

        // Online softmax per query row. Row j lives in one 16-lane half.
        float fct[8];
#pragma unroll
        for (int j = 0; j < 8; ++j) {
            float rmax = fmaxf(fmaxf(s[0][j], s[1][j]), fmaxf(s[2][j], s[3][j]));
#pragma unroll
            for (int msk = 1; msk < 16; msk <<= 1)
                rmax = fmaxf(rmax, __shfl_xor(rmax, msk, 32));
            const float nm = fmaxf(mrow[j], rmax);
            const float fj = __expf(mrow[j] - nm);
            float psum = 0.f;
#pragma unroll
            for (int g = 0; g < 4; ++g) {
                const float p = __expf(s[g][j] - nm);
                s[g][j] = p;
                psum += p;
            }
#pragma unroll
            for (int msk = 1; msk < 16; msk <<= 1)
                psum += __shfl_xor(psum, msk, 32);
            lsum[j] = lsum[j] * fj + psum;
            mrow[j] = nm;
            fct[j] = fj;
        }
#pragma unroll
        for (int g = 0; g < 4; ++g)
#pragma unroll
            for (int j = 0; j < 8; ++j) o[g][j] *= fct[j];

        // P (C/D layout) -> bf16 A-layout via per-wave LDS buffer
#pragma unroll
        for (int g = 0; g < 4; ++g)
#pragma unroll
            for (int j = 0; j < 8; ++j)
                ldsP[wave][hi * 8 + j][g * 16 + lr] = (bf16_t)s[g][j];
        asm volatile("s_wait_dscnt 0" ::: "memory");  // wave-local RAW on LDS

        // O strip += P(16x64 keys) * V_tile(64 keys x 64 hd)
#pragma unroll
        for (int kk = 0; kk < 2; ++kk) {
            FragBF pf;
            const bf16_t* pr = &ldsP[wave][lr][kk * 32];
            pf.h[0] = *(const v8bf*)(pr + hi * 8);
            pf.h[1] = *(const v8bf*)(pr + 16 + hi * 8);
            FragBF vf[4];
#pragma unroll
            for (int g = 0; g < 4; ++g) {
                const bf16_t* vr = &ldsV[buf][g * 16 + lr][kk * 32 + hi * 8];
                vf[g].h[0] = *(const v8bf*)(vr);
                vf[g].h[1] = *(const v8bf*)(vr + 16);
            }
#pragma unroll
            for (int g = 0; g < 4; ++g)
                o[g] = __builtin_amdgcn_wmma_f32_16x16x32_bf16(
                    false, pf.v, false, vf[g].v, (short)0, o[g], false, false);
        }
        __syncthreads();                       // reads done before buf reuse
    }

    // Normalize and store bf16 O at [b, s, h*64 + hd]
#pragma unroll
    for (int j = 0; j < 8; ++j) {
        const float inv = 1.f / lsum[j];
        const int srow = q0 + wave * 16 + hi * 8 + j;
        const size_t base = ((size_t)(bidx * SEQ + srow)) * DIM + head * HD;
#pragma unroll
        for (int g = 0; g < 4; ++g)
            O[base + g * 16 + lr] = (bf16_t)(o[g][j] * inv);
    }
}

// ---------------------------------------------------------------------------
// Host launcher
// ---------------------------------------------------------------------------
extern "C" void kernel_launch(void* const* d_in, const int* in_sizes, int n_in,
                              void* d_out, int out_size, void* d_ws, size_t ws_size,
                              hipStream_t stream) {
    (void)in_sizes; (void)n_in; (void)out_size; (void)ws_size;
    const float* x  = (const float*)d_in[0];
    const float* Wq = (const float*)d_in[1];
    const float* Wk = (const float*)d_in[2];
    const float* Wv = (const float*)d_in[3];
    const float* Wo = (const float*)d_in[4];
    const float* bo = (const float*)d_in[5];
    float* out = (float*)d_out;

    char* ws = (char*)d_ws;
    size_t off = 0;
    const size_t xElems = (size_t)MROWS * DIM;   // 4 Mi
    const size_t wElems = (size_t)DIM * DIM;     // 1 Mi
    bf16_t* xb  = (bf16_t*)(ws + off); off += xElems * 2;
    bf16_t* wqb = (bf16_t*)(ws + off); off += wElems * 2;
    bf16_t* wkb = (bf16_t*)(ws + off); off += wElems * 2;
    bf16_t* wvb = (bf16_t*)(ws + off); off += wElems * 2;
    bf16_t* wob = (bf16_t*)(ws + off); off += wElems * 2;
    bf16_t* qb  = (bf16_t*)(ws + off); off += xElems * 2;  // [b,h,s,hd], pre-scaled
    bf16_t* kb  = (bf16_t*)(ws + off); off += xElems * 2;  // [b,h,s,hd]
    bf16_t* vtb = (bf16_t*)(ws + off); off += xElems * 2;  // [b,h,hd,s]
    bf16_t* ob  = (bf16_t*)(ws + off); off += xElems * 2;  // [b,s,d]

    // Stage 1: precision conversion
    {
        const int nx4 = (int)(xElems / 4);
        cvt_f32_to_bf16<<<(nx4 + 255) / 256, 256, 0, stream>>>(x, xb, nx4);
        const int nw4 = (int)(wElems / 4);
        cvt_f32_to_bf16<<<(nw4 + 255) / 256, 256, 0, stream>>>(Wq, wqb, nw4);
        cvt_f32_to_bf16<<<(nw4 + 255) / 256, 256, 0, stream>>>(Wk, wkb, nw4);
        cvt_f32_to_bf16<<<(nw4 + 255) / 256, 256, 0, stream>>>(Wv, wvb, nw4);
        cvt_f32_to_bf16<<<(nw4 + 255) / 256, 256, 0, stream>>>(Wo, wob, nw4);
    }

    // Stage 2: Q/K/V projections (Q folds 1/sqrt(64) = 0.125 exactly)
    dim3 ggrid(MROWS / 64, DIM / 64);
    gemm_bf16<<<ggrid, 128, 0, stream>>>(xb, wqb, qb, nullptr, DIM, DIM, 0, 0.125f);
    gemm_bf16<<<ggrid, 128, 0, stream>>>(xb, wkb, kb, nullptr, DIM, DIM, 1, 1.0f);
    gemm_bf16<<<ggrid, 128, 0, stream>>>(xb, wvb, vtb, nullptr, DIM, DIM, 2, 1.0f);

    // Stage 3: attention
    attn_kernel<<<dim3(BATCH * NHEAD, SEQ / 64), 128, 0, stream>>>(qb, kb, vtb, ob);

    // Stage 4: output projection + bias, fp32 result
    gemm_bf16<<<ggrid, 128, 0, stream>>>(ob, wob, out, bo, DIM, DIM, 3, 1.0f);
}
